// chemTemplateCpLayer_9947144257543
// MI455X (gfx1250) — matrix operand order repaired and checked by reading, not compiled, not run on hardware
//
#include <hip/hip_runtime.h>

// ---------------------------------------------------------------------------
// Chemical-template network fixed-point solver, MI455X (gfx1250, wave32, WMMA)
//
//   L=4 layers, N=1024 nodes, B=256 samples, 20 bisection iters (+born_sup).
//   Dominant work: 21 * 4 * 2 GEMMs (1024x1024)@(1024x256) ~= 90 GFLOP.
//   Prepped f16 Cact/Cinh (16 MB) live in the 192 MB L2; HBM cost is one
//   ~150 MB prep sweep (~6.5 us @ 23.3 TB/s). Core loop = v_wmma f16->f32.
//   Single 32 KB LDS X buffer (in-place update behind a barrier) keeps the
//   dynamic-LDS request at ~52 KB, under conservative runtime caps.
// ---------------------------------------------------------------------------

typedef _Float16 v16h __attribute__((ext_vector_type(16)));
typedef _Float16 v8h  __attribute__((ext_vector_type(8)));
typedef float    v8f  __attribute__((ext_vector_type(8)));
typedef float    v4f  __attribute__((ext_vector_type(4)));

#define L_LAYERS 4
#define N_DIM    1024
#define B_BATCH  256
#define BT       16     // batch columns per block (WMMA N)
#define N_BIS    20

// ---------------------------------------------------------------------------
// Prep 1: per-edge effective constants, masked, cast to f16 for WMMA A-matrices
// ---------------------------------------------------------------------------
__global__ void prep_mats(const float* __restrict__ k1, const float* __restrict__ k1n,
                          const float* __restrict__ k2, const float* __restrict__ TA0,
                          const float* __restrict__ Cinhib0, const float* __restrict__ masks,
                          const float* __restrict__ E0,
                          _Float16* __restrict__ cact, _Float16* __restrict__ cinh) {
  const int idx = blockIdx.x * blockDim.x + threadIdx.x;   // grid covers L*N*N exactly
  const float m  = masks[idx];
  const float e0 = E0[0];
  float ca = 0.f, ci = 0.f;
  if (m > 0.f) ca = k2[idx] * k1[idx] * TA0[idx] * e0 / (k1n[idx] + k2[idx]);
  if (m < 0.f) ci = Cinhib0[idx];
  cact[idx] = (_Float16)ca;
  cinh[idx] = (_Float16)ci;
}

// ---------------------------------------------------------------------------
// Prep 2: w[l][in] = sum_out (Kactiv + Kinhib)   (column sums, coalesced in `in`)
// ---------------------------------------------------------------------------
__global__ void prep_w(const float* __restrict__ k1, const float* __restrict__ k1n,
                       const float* __restrict__ k2, const float* __restrict__ k3,
                       const float* __restrict__ k3n, const float* __restrict__ k4,
                       const float* __restrict__ TA0, const float* __restrict__ TI0,
                       const float* __restrict__ masks, float* __restrict__ wOut) {
  const int tid = blockIdx.x * blockDim.x + threadIdx.x;   // L*N threads
  const int l = tid >> 10, in = tid & (N_DIM - 1);
  float s = 0.f;
  for (int out = 0; out < N_DIM; ++out) {
    const size_t i = ((size_t)l * N_DIM + out) * N_DIM + in;
    const float m = masks[i];
    if (m > 0.f) s += k1[i] * TA0[i] / (k1n[i] + k2[i]);
    if (m < 0.f) s += k3[i] * TI0[i] / (k3n[i] + k4[i]);
  }
  wOut[tid] = s;
}

// ---------------------------------------------------------------------------
// Main persistent kernel: one block per 16-sample tile, 1024 threads = 32 waves.
// ---------------------------------------------------------------------------
__global__ void __launch_bounds__(1024, 1)
bisect_kernel(const float* __restrict__ inputs,
              const _Float16* __restrict__ cact,
              const _Float16* __restrict__ cinh,
              const float* __restrict__ wArr,
              const float* __restrict__ k6g,
              const float* __restrict__ kdIg,
              const float* __restrict__ kdTg,
              float* __restrict__ out) {
  extern __shared__ char smem[];
  _Float16* Xt     = (_Float16*)smem;                         // [BT][N_DIM] f16, 32 KB
  float*    ws     = (float*)(smem + BT * N_DIM * sizeof(_Float16)); // [L][N] 16 KB
  float*    partial= ws + L_LAYERS * N_DIM;                   // [BT][64]  4 KB
  float*    loA    = partial + BT * 64;
  float*    hiA    = loA + BT;
  float*    cpA    = hiA + BT;
  float*    GA     = cpA + BT;

  const int t    = threadIdx.x;
  const int lane = t & 31;
  const int wv   = t >> 5;               // 0..31
  const int colBase = blockIdx.x * BT;

  // preload w into LDS
  for (int i = t; i < L_LAYERS * N_DIM; i += 1024) ws[i] = wArr[i];
  __syncthreads();

  // WMMA lane addressing (CDNA5 ISA 7.12.2, wave32)
  const int h        = lane >> 4;        // lane half
  const int aRowLane = lane & 15;        // A: row M = lane%16
  const int aKoff    = 16 * h;           // A: K 16h..16h+15 (contiguous, matches B)
  const int bCol     = lane & 15;        // B/D: column N = lane%16
  const int bKoff    = 16 * h;           // B: K 0-15 (lanes 0-15), 16-31 (16-31)

  for (int ev = 0; ev <= N_BIS; ++ev) {
    const bool bornsup = (ev == 0);
    if (t < BT) {
      cpA[t] = bornsup ? 1.0f : 0.5f * (loA[t] + hiA[t]);
      GA[t]  = 1.0f;                     // G = 1 + sum(dcp)
    }
    // X0 = inputs (same for every eval), f16, transposed into Xt
    for (int i = t; i < BT * N_DIM; i += 1024) {
      const int col = i >> 10, k = i & (N_DIM - 1);
      Xt[col * N_DIM + k] = (_Float16)inputs[(size_t)(colBase + col) * N_DIM + k];
    }
    __syncthreads();

    for (int l = 0; l < L_LAYERS; ++l) {
      const int lN = l * N_DIM;

      // ---- dcp_l[col] = sum_in w[l][in] * X[in][col]  (cheap VALU side job)
      {
        const int col = t & 15, chunk = t >> 4;       // 64 chunks of 16
        const float* wl = ws + lN;
        const int base = chunk * 16;
        float s = 0.f;
#pragma unroll
        for (int j = 0; j < 16; ++j)
          s += wl[base + j] * (float)Xt[col * N_DIM + base + j];
        partial[col * 64 + chunk] = s;
      }
      __syncthreads();
      if (t < BT) {
        float s = 0.f;
#pragma unroll 8
        for (int c = 0; c < 64; ++c) s += partial[t * 64 + c];
        GA[t] += s;
      }

      // ---- Y = Cact@X, Z = Cinh@X via WMMA; 64 row-tiles over 32 waves,
      //      results staged in registers so X can be updated in place.
      const float cpn = cpA[bCol];
      v8h xout[2];
#pragma unroll
      for (int rep = 0; rep < 2; ++rep) {
        const int m0 = 16 * (2 * wv + rep);
        // wave-uniform panel base + lane-varying 32-bit offset (GVS addressing)
        const _Float16* __restrict__ aPanY = cact + ((size_t)lN + m0) * N_DIM;
        const _Float16* __restrict__ aPanZ = cinh + ((size_t)lN + m0) * N_DIM;
        const int aLane = aRowLane * N_DIM + aKoff;   // + kb per step
        v8f accY = {}; v8f accZ = {};
#pragma unroll 4
        for (int kb = 0; kb < N_DIM; kb += 32) {
          const int ao = aLane + kb;
          v8h aY0 = *(const v8h*)(aPanY + ao);
          v8h aY1 = *(const v8h*)(aPanY + ao + 8);
          v8h aZ0 = *(const v8h*)(aPanZ + ao);
          v8h aZ1 = *(const v8h*)(aPanZ + ao + 8);
          v16h Bm = *(const v16h*)(Xt + bCol * N_DIM + kb + bKoff);  // 32B LDS load
          v16h AY = __builtin_shufflevector(aY0, aY1, 0,1,2,3,4,5,6,7,8,9,10,11,12,13,14,15);
          v16h AZ = __builtin_shufflevector(aZ0, aZ1, 0,1,2,3,4,5,6,7,8,9,10,11,12,13,14,15);
          accY = __builtin_amdgcn_wmma_f32_16x16x32_f16(false, AY, false, Bm,
                                                        (short)0, accY, false, false);
          accZ = __builtin_amdgcn_wmma_f32_16x16x32_f16(false, AZ, false, Bm,
                                                        (short)0, accZ, false, false);
        }
        // epilogue: D layout -> lane holds col bCol, rows m0+8h+v (v=0..7)
        const int mBase = m0 + 8 * h;
        const float* kdTrow = kdTg + lN + mBase;      // L2-resident, 32B-aligned
        v4f kt0 = *(const v4f*)(kdTrow);
        v4f kt1 = *(const v4f*)(kdTrow + 4);
        v8h xo;
        if (bornsup) {
#pragma unroll
          for (int v = 0; v < 8; ++v)
            xo[v] = (_Float16)(accY[v] / (v < 4 ? kt0[v] : kt1[v - 4]));
        } else {
          const float* kdIrow = kdIg + lN + mBase;
          const float* k6row  = k6g  + lN + mBase;
          v4f ki0 = *(const v4f*)(kdIrow);
          v4f ki1 = *(const v4f*)(kdIrow + 4);
          v4f k60 = *(const v4f*)(k6row);
          v4f k61 = *(const v4f*)(k6row + 4);
#pragma unroll
          for (int v = 0; v < 8; ++v) {
            const float kdTm = (v < 4) ? kt0[v] : kt1[v - 4];
            const float kdIm = (v < 4) ? ki0[v] : ki1[v - 4];
            const float k6m  = (v < 4) ? k60[v] : k61[v - 4];
            const float inv = 1.0f / (kdTm * cpn);
            const float xe  = accY[v] * inv;
            const float ih  = accZ[v] * inv;
            xo[v] = (_Float16)(xe / (1.0f + k6m * ih / (kdIm * cpn)));
          }
        }
        xout[rep] = xo;
      }
      __syncthreads();                   // all reads of Xt (dcp + both GEMMs) done
#pragma unroll
      for (int rep = 0; rep < 2; ++rep) {
        const int mBase = 16 * (2 * wv + rep) + 8 * h;
        *(v8h*)(Xt + bCol * N_DIM + mBase) = xout[rep];   // in-place 16B LDS store
      }
      __syncthreads();
    }

    // ---- bisection bookkeeping (G is decreasing in cp)
    if (t < BT) {
      const float G = GA[t];
      if (bornsup)               { loA[t] = 1.0f; hiA[t] = G; }
      else if (G - cpA[t] > 0.f) loA[t] = cpA[t];
      else                       hiA[t] = cpA[t];
    }
    __syncthreads();
  }

  if (t < BT) out[colBase + t] = 0.5f * (loA[t] + hiA[t]);
}

// ---------------------------------------------------------------------------
extern "C" void kernel_launch(void* const* d_in, const int* in_sizes, int n_in,
                              void* d_out, int out_size, void* d_ws, size_t ws_size,
                              hipStream_t stream) {
  (void)in_sizes; (void)n_in; (void)out_size; (void)ws_size;
  const float* inputs  = (const float*)d_in[0];
  const float* k1      = (const float*)d_in[1];
  const float* k1n     = (const float*)d_in[2];
  const float* k2      = (const float*)d_in[3];
  const float* k3      = (const float*)d_in[4];
  const float* k3n     = (const float*)d_in[5];
  const float* k4      = (const float*)d_in[6];
  const float* k6      = (const float*)d_in[7];
  const float* kdI     = (const float*)d_in[8];
  const float* kdT     = (const float*)d_in[9];
  const float* TA0     = (const float*)d_in[10];
  const float* TI0     = (const float*)d_in[11];
  const float* Cinhib0 = (const float*)d_in[12];
  const float* E0      = (const float*)d_in[13];
  const float* masks   = (const float*)d_in[14];
  float* out = (float*)d_out;

  char* ws = (char*)d_ws;
  const size_t mat = (size_t)L_LAYERS * N_DIM * N_DIM;   // 4M elements
  _Float16* cact = (_Float16*)ws;                        // 8 MB
  _Float16* cinh = (_Float16*)(ws + mat * sizeof(_Float16));
  float*    wArr = (float*)(ws + 2 * mat * sizeof(_Float16)); // 16 KB

  prep_mats<<<dim3((unsigned)(mat / 256)), dim3(256), 0, stream>>>(
      k1, k1n, k2, TA0, Cinhib0, masks, E0, cact, cinh);
  prep_w<<<dim3((L_LAYERS * N_DIM) / 256), dim3(256), 0, stream>>>(
      k1, k1n, k2, k3, k3n, k4, TA0, TI0, masks, wArr);

  const size_t lds = BT * N_DIM * sizeof(_Float16)          // X buffer: 32 KB
                   + L_LAYERS * N_DIM * sizeof(float)       // w:        16 KB
                   + BT * 64 * sizeof(float)                // partials:  4 KB
                   + 4 * BT * sizeof(float);                // lo/hi/cp/G
  bisect_kernel<<<dim3(B_BATCH / BT), dim3(1024), lds, stream>>>(
      inputs, cact, cinh, wArr, k6, kdI, kdT, out);
}
